// TransformerModel_17617955848985
// MI455X (gfx1250) — matrix operand order repaired and checked by reading, not compiled, not run on hardware
//
#include <hip/hip_runtime.h>
#include <hip/hip_bf16.h>
#include <math.h>

// ---------------------------------------------------------------------------
// Model constants (match reference)
// ---------------------------------------------------------------------------
namespace {
constexpr int   kSeg    = 32;      // tokens per ray
constexpr int   kDim    = 128;
constexpr int   kFF     = 512;
constexpr int   kHeads  = 4;
constexpr int   kDHead  = 32;
constexpr float kRadius = 3.0f;
constexpr float kSegLen = 2.0f * kRadius / kSeg;   // 0.1875
constexpr int   kEnc    = 72;
constexpr int   kEncPad = 96;      // pad K to multiple of 32 for WMMA

// f16 workspace layout (element offsets), weights pre-tiled to B-fragment order
constexpr size_t kWupSz   = 3 * 8 * 512;           // KT=3, NT=8
constexpr size_t kLayerSz = 4 * (4 * 8 * 512) + 2 * (4 * 32 * 512); // 196608
constexpr size_t kWqOff = 0;
constexpr size_t kWkOff = 16384;
constexpr size_t kWvOff = 32768;
constexpr size_t kWoOff = 49152;
constexpr size_t kW1Off = 65536;   // KT=4, NT=32 -> 65536
constexpr size_t kW2Off = 131072;  // KT=16, NT=8 -> 65536
}

typedef _Float16 v16h __attribute__((ext_vector_type(16)));
typedef float    v8f  __attribute__((ext_vector_type(8)));

// ---------------------------------------------------------------------------
// Weight conversion: fp32 row-major [K][N] -> f16 WMMA-B fragment tiles.
// dst layout: [kt][nt][lane][16], lane n = nt*16+(lane&15), k = kt*32+16*(lane>>4)+p
// ---------------------------------------------------------------------------
__global__ void nerftx_convert_kernel(const float* __restrict__ W,
                                      _Float16* __restrict__ dst,
                                      int K, int N, int Kpad) {
  const int total = (Kpad >> 5) * (N >> 4) * 512;
  int idx = blockIdx.x * blockDim.x + threadIdx.x;
  if (idx >= total) return;
  const int p    = idx & 15;
  const int lane = (idx >> 4) & 31;
  const int tile = idx >> 9;
  const int NT   = N >> 4;
  const int nt   = tile % NT;
  const int kt   = tile / NT;
  const int k    = kt * 32 + ((lane >> 4) << 4) + p;
  const int n    = nt * 16 + (lane & 15);
  float v = (k < K) ? W[(size_t)k * N + n] : 0.0f;
  dst[idx] = (_Float16)v;
}

// ---------------------------------------------------------------------------
// WMMA helpers (CDNA5 16x16x32 f16 -> f32)
// ---------------------------------------------------------------------------
__device__ __forceinline__ v8f wmma_f16(v16h a, v16h b, v8f c) {
  return __builtin_amdgcn_wmma_f32_16x16x32_f16(false, a, false, b, (short)0, c,
                                                false, false);
}

// A fragment from LDS f16 row-major [M][lda]: contiguous k = 16*khalf + s.
__device__ __forceinline__ v16h load_a_frag(const _Float16* A, int lda,
                                            int mbase, int kbase, int lane) {
  const int m  = mbase + (lane & 15);
  const int kh = (lane >> 4) << 4;   // 16*khalf
  const _Float16* row = A + m * lda + kbase;
  union { v16h v; unsigned int u[8]; } r;
#pragma unroll
  for (int g = 0; g < 8; ++g) {
    const int k0 = kh + 2 * g;
    r.u[g] = *(const unsigned int*)(row + k0);
  }
  return r.v;
}

// C = A(LDS f16, 32 x 32*KT) @ B(tiled global f16) [+bias][relu]
// outmode: 0 -> store f16 to C16, 1 -> store f32 to C32, 2 -> += into C32
__device__ void gemm32(const _Float16* __restrict__ A, int lda,
                       const _Float16* __restrict__ Bt, int KT, int NT,
                       const float* __restrict__ bias, int outmode, bool relu,
                       _Float16* __restrict__ C16, float* __restrict__ C32,
                       int ldc, int wave, int lane) {
  const int ntpw = NT >> 2;          // tiles in N per wave (4 waves)
  for (int mt = 0; mt < 2; ++mt) {
    for (int j = 0; j < ntpw; ++j) {
      const int nt = wave * ntpw + j;
      v8f acc;
#pragma unroll
      for (int g = 0; g < 8; ++g) acc[g] = 0.0f;
      for (int kt = 0; kt < KT; ++kt) {
        v16h a = load_a_frag(A, lda, mt * 16, kt * 32, lane);
        v16h b = *(const v16h*)(Bt + ((size_t)((kt * NT + nt) * 32 + lane) << 4));
        acc = wmma_f16(a, b, acc);
      }
      const int   n  = nt * 16 + (lane & 15);
      const float bv = bias ? bias[n] : 0.0f;
      const int   mb = mt * 16 + ((lane >> 4) << 3);
#pragma unroll
      for (int g = 0; g < 8; ++g) {
        float v = acc[g] + bv;
        if (relu) v = fmaxf(v, 0.0f);
        const int m = mb + g;
        if (outmode == 0)      C16[m * ldc + n] = (_Float16)v;
        else if (outmode == 1) C32[m * ldc + n] = v;
        else                   C32[m * ldc + n] += v;
      }
    }
  }
}

// ---------------------------------------------------------------------------
// Shared-memory overlays
// ---------------------------------------------------------------------------
struct AttnBufs {
  _Float16 Q[kSeg][kDim];
  _Float16 Kt[kSeg][kDim];
  _Float16 V[kSeg][kDim];
  _Float16 S[kHeads][kSeg][kSeg];   // scores, then softmaxed in place
};
union SharedU {
  _Float16 enc[kSeg][kEncPad];      // 6 KB
  AttnBufs at;                      // 32 KB
  _Float16 U[kSeg][kFF];            // 32 KB (FFN hidden)
};

// LayerNorm over last dim (128), X fp32 -> H f16, 4 threads per row.
__device__ void layernorm128(const float X[][kDim], _Float16 H[][kDim],
                             const float* __restrict__ gamma,
                             const float* __restrict__ beta,
                             float* red_s, float* red_q,
                             float* muv, float* rsv, int tid) {
  const int r = tid >> 2, q = tid & 3;
  float s = 0.0f, ss = 0.0f;
  for (int i = 0; i < 32; ++i) {
    float v = X[r][q * 32 + i];
    s += v; ss += v * v;
  }
  red_s[r * 4 + q] = s;
  red_q[r * 4 + q] = ss;
  __syncthreads();
  if (tid < 32) {
    float su = red_s[tid*4] + red_s[tid*4+1] + red_s[tid*4+2] + red_s[tid*4+3];
    float sq = red_q[tid*4] + red_q[tid*4+1] + red_q[tid*4+2] + red_q[tid*4+3];
    float mu  = su / (float)kDim;
    float var = sq / (float)kDim - mu * mu;
    muv[tid] = mu;
    rsv[tid] = rsqrtf(var + 1e-5f);
  }
  __syncthreads();
  {
    const float mu = muv[r], rs = rsv[r];
    for (int i = 0; i < 32; ++i) {
      const int d = q * 32 + i;
      H[r][d] = (_Float16)((X[r][d] - mu) * rs * gamma[d] + beta[d]);
    }
  }
  __syncthreads();
}

// ---------------------------------------------------------------------------
// Main kernel: one ray per 128-thread block (4 wave32s)
// ---------------------------------------------------------------------------
__global__ __launch_bounds__(128)
void nerftx_ray_kernel(const float* __restrict__ points,
                       const _Float16* __restrict__ wst,
                       const float* __restrict__ b_up,
                       const float* __restrict__ ln1_s, const float* __restrict__ ln1_b,
                       const float* __restrict__ ln2_s, const float* __restrict__ ln2_b,
                       const float* __restrict__ b1,   const float* __restrict__ b2,
                       const float* __restrict__ w_pool,
                       const float* __restrict__ W_hit, const float* __restrict__ b_hit,
                       const float* __restrict__ W_dc,  const float* __restrict__ b_dc,
                       const float* __restrict__ W_dv,  const float* __restrict__ b_dv,
                       float* __restrict__ out, int n_rays) {
  __shared__ __align__(32) float   Xf32[kSeg][kDim];   // 16 KB activations (fp32)
  __shared__ __align__(32) _Float16 Hh[kSeg][kDim];    // 8 KB  LN output / attn O (f16)
  __shared__ __align__(32) SharedU sh;                 // 32 KB union
  __shared__ float g_orig[3], g_vec[3], g_t1;
  __shared__ int   g_nl;
  __shared__ float ln_s[128], ln_q[128], ln_mu[32], ln_rs[32];
  __shared__ float lgts[32], clsa[32], vala[32], red[128];

  const int ray  = blockIdx.x;
  const int tid  = threadIdx.x;
  const int wave = tid >> 5;
  const int lane = tid & 31;
  if (ray >= n_rays) return;

  // ---- ray/sphere geometry (thread 0) ----
  if (tid == 0) {
    const float* p = points + (size_t)ray * 6;
    float ox = p[0], oy = p[1], oz = p[2];
    float vx = p[3] - ox, vy = p[4] - oy, vz = p[5] - oz;
    float inv = rsqrtf(vx * vx + vy * vy + vz * vz);
    vx *= inv; vy *= inv; vz *= inv;
    float bq = ox * vx + oy * vy + oz * vz;
    float cq = ox * ox + oy * oy + oz * oz - kRadius * kRadius;
    float sq = sqrtf(fmaxf(bq * bq - cq, 0.0f));
    float t1 = -bq - sq, t2 = -bq + sq;
    int nl = 0;
    for (int j = 0; j < kSeg; ++j)
      if (t1 + (float)j * kSegLen < t2) ++nl;
    g_orig[0] = ox; g_orig[1] = oy; g_orig[2] = oz;
    g_vec[0] = vx; g_vec[1] = vy; g_vec[2] = vz;
    g_t1 = t1; g_nl = nl;
  }
  __syncthreads();

  // ---- positional encoding -> sh.enc[32][96] f16 (K padded with zeros) ----
  {
    const int  seg  = tid >> 2, q = tid & 3;
    const bool keep = seg < g_nl;
    const float t0 = g_t1 + (float)seg * kSegLen;
    const float pi = 3.14159265358979323846f;
    for (int ii = 0; ii < 24; ++ii) {
      const int idx = q * 24 + ii;
      _Float16 val;
      if (idx < kEnc) {
        const int e = idx / 36;
        const int rem = idx - e * 36;
        const int c = rem / 12;
        const int j = rem - c * 12;
        const float tt = t0 + (float)e * kSegLen;
        const float coord = keep ? ((g_orig[c] + g_vec[c] * tt) / kRadius) : 0.0f;
        const int f = (j < 6) ? j : (j - 6);
        const float ang = coord * (pi * (float)(1 << f));
        val = (_Float16)((j < 6) ? sinf(ang) : cosf(ang));
      } else {
        val = (_Float16)0.0f;
      }
      sh.enc[seg][idx] = val;
    }
  }
  __syncthreads();

  // ---- x = enc @ W_up + b_up (32x96 @ 96x128, fp32 out) ----
  gemm32(&sh.enc[0][0], kEncPad, wst, 3, 8, b_up, /*out*/1, false,
         nullptr, &Xf32[0][0], kDim, wave, lane);
  __syncthreads();

  const float ascale = 0.1767766952966369f;  // 1/sqrt(D_HEAD)

  for (int L = 0; L < 2; ++L) {
    const _Float16* lw = wst + kWupSz + (size_t)L * kLayerSz;

    // LN1 -> Hh (f16)
    layernorm128(Xf32, Hh, ln1_s + L * kDim, ln1_b + L * kDim,
                 ln_s, ln_q, ln_mu, ln_rs, tid);

    // Q/K/V projections (write disjoint union regions, all read Hh)
    gemm32(&Hh[0][0], kDim, lw + kWqOff, 4, 8, nullptr, 0, false,
           &sh.at.Q[0][0], nullptr, kDim, wave, lane);
    gemm32(&Hh[0][0], kDim, lw + kWkOff, 4, 8, nullptr, 0, false,
           &sh.at.Kt[0][0], nullptr, kDim, wave, lane);
    gemm32(&Hh[0][0], kDim, lw + kWvOff, 4, 8, nullptr, 0, false,
           &sh.at.V[0][0], nullptr, kDim, wave, lane);
    __syncthreads();

    // ---- attention scores: S_h = scale * Q_h @ K_h^T  (wave == head) ----
    {
      const int h = wave;
      for (int mt = 0; mt < 2; ++mt) {
        for (int nt = 0; nt < 2; ++nt) {
          v8f acc;
#pragma unroll
          for (int g = 0; g < 8; ++g) acc[g] = 0.0f;
          v16h a = load_a_frag(&sh.at.Q[0][0], kDim, mt * 16, h * kDHead, lane);
          // B = K^T: contiguous 16 f16 per lane along d
          v16h b = *(const v16h*)(&sh.at.Kt[nt * 16 + (lane & 15)]
                                          [h * kDHead + ((lane >> 4) << 4)]);
          acc = wmma_f16(a, b, acc);
          const int n  = nt * 16 + (lane & 15);
          const int mb = mt * 16 + ((lane >> 4) << 3);
#pragma unroll
          for (int g = 0; g < 8; ++g)
            sh.at.S[h][mb + g][n] = (_Float16)(acc[g] * ascale);
        }
      }
    }
    __syncthreads();

    // ---- softmax rows (one thread per (head,row)) ----
    {
      const int h2 = tid >> 5, r = tid & 31;
      float row[32];
      float mx = -1e30f;
      for (int j = 0; j < 32; ++j) {
        row[j] = (float)sh.at.S[h2][r][j];
        mx = fmaxf(mx, row[j]);
      }
      float den = 0.0f;
      for (int j = 0; j < 32; ++j) { row[j] = expf(row[j] - mx); den += row[j]; }
      const float inv = 1.0f / den;
      for (int j = 0; j < 32; ++j)
        sh.at.S[h2][r][j] = (_Float16)(row[j] * inv);
    }
    __syncthreads();

    // ---- O_h = att_h @ V_h -> Hh (f16, head-major columns) ----
    {
      const int h = wave;
      for (int mt = 0; mt < 2; ++mt) {
        for (int nt = 0; nt < 2; ++nt) {
          v8f acc;
#pragma unroll
          for (int g = 0; g < 8; ++g) acc[g] = 0.0f;
          v16h a = load_a_frag(&sh.at.S[h][0][0], 32, mt * 16, 0, lane);
          union { v16h v; _Float16 e[16]; } bb;
          const int kh  = (lane >> 4) << 4;
          const int col = h * kDHead + nt * 16 + (lane & 15);
#pragma unroll
          for (int p = 0; p < 16; ++p) bb.e[p] = sh.at.V[kh + p][col];
          acc = wmma_f16(a, bb.v, acc);
          const int n  = nt * 16 + (lane & 15);
          const int mb = mt * 16 + ((lane >> 4) << 3);
#pragma unroll
          for (int g = 0; g < 8; ++g)
            Hh[mb + g][h * kDHead + n] = (_Float16)acc[g];
        }
      }
    }
    __syncthreads();

    // x += O @ Wo
    gemm32(&Hh[0][0], kDim, lw + kWoOff, 4, 8, nullptr, /*accum*/2, false,
           nullptr, &Xf32[0][0], kDim, wave, lane);
    __syncthreads();

    // LN2 -> Hh
    layernorm128(Xf32, Hh, ln2_s + L * kDim, ln2_b + L * kDim,
                 ln_s, ln_q, ln_mu, ln_rs, tid);

    // U = relu(H @ W1 + b1) (f16, 32x512)
    gemm32(&Hh[0][0], kDim, lw + kW1Off, 4, 32, b1 + L * kFF, 0, true,
           &sh.U[0][0], nullptr, kFF, wave, lane);
    __syncthreads();

    // x += U @ W2 + b2
    gemm32(&sh.U[0][0], kFF, lw + kW2Off, 16, 8, b2 + L * kDim, 2, false,
           nullptr, &Xf32[0][0], kDim, wave, lane);
    __syncthreads();
  }

  // ---- heads: pooling logits, dist_cls, dist_val ----
  if (tid < 96) {
    const int which = tid >> 5, s = tid & 31;
    const float* w = (which == 0) ? w_pool : (which == 1) ? W_dc : W_dv;
    float acc = 0.0f;
    for (int d = 0; d < kDim; ++d) acc += Xf32[s][d] * w[d];
    if (which == 0)      lgts[s] = acc;
    else if (which == 1) clsa[s] = acc + b_dc[0];
    else                 vala[s] = fminf(fmaxf(acc + b_dv[0], 0.0f), 1.0f);
  }
  __syncthreads();

  // pooled . W_hit partials (one dim per thread)
  {
    float mx = -1e30f;
    for (int s = 0; s < kSeg; ++s) mx = fmaxf(mx, lgts[s]);
    float asv[kSeg];
    float den = 0.0f;
    for (int s = 0; s < kSeg; ++s) { asv[s] = expf(lgts[s] - mx); den += asv[s]; }
    const float inv = 1.0f / den;
    const int d = tid;
    float pd = 0.0f;
    for (int s = 0; s < kSeg; ++s) pd += asv[s] * Xf32[s][d];
    red[tid] = pd * inv * W_hit[d];
  }
  __syncthreads();

  if (tid == 0) {
    float hit = b_hit[0];
    for (int t = 0; t < 128; ++t) hit += red[t];
    if (g_nl == 0) hit = -100.0f;
    float best = -INFINITY;
    int bi = 0;
    for (int s = 0; s < kSeg; ++s) {
      const float c = (s < g_nl) ? clsa[s] : -INFINITY;
      if (c > best) { best = c; bi = s; }
    }
    const float dist = vala[bi] + (float)bi * kSegLen + g_t1;
    out[ray] = hit;
    out[n_rays + ray] = dist;
  }
}

// ---------------------------------------------------------------------------
// Host launcher
// ---------------------------------------------------------------------------
extern "C" void kernel_launch(void* const* d_in, const int* in_sizes, int n_in,
                              void* d_out, int out_size, void* d_ws, size_t ws_size,
                              hipStream_t stream) {
  const float* points = (const float*)d_in[0];
  const float* W_up   = (const float*)d_in[1];
  const float* b_up   = (const float*)d_in[2];
  const float* ln1_s  = (const float*)d_in[3];
  const float* ln1_b  = (const float*)d_in[4];
  const float* Wq     = (const float*)d_in[5];
  const float* Wk     = (const float*)d_in[6];
  const float* Wv     = (const float*)d_in[7];
  const float* Wo     = (const float*)d_in[8];
  const float* ln2_s  = (const float*)d_in[9];
  const float* ln2_b  = (const float*)d_in[10];
  const float* W1     = (const float*)d_in[11];
  const float* b1     = (const float*)d_in[12];
  const float* W2     = (const float*)d_in[13];
  const float* b2     = (const float*)d_in[14];
  const float* w_pool = (const float*)d_in[15];
  const float* W_hit  = (const float*)d_in[16];
  const float* b_hit  = (const float*)d_in[17];
  const float* W_dc   = (const float*)d_in[18];
  const float* b_dc   = (const float*)d_in[19];
  const float* W_dv   = (const float*)d_in[20];
  const float* b_dv   = (const float*)d_in[21];

  _Float16* ws = (_Float16*)d_ws;
  const int n_rays = in_sizes[0] / 6;

  auto conv = [&](const float* W, size_t dstoff, int K, int N, int Kpad) {
    const int total = (Kpad / 32) * (N / 16) * 512;
    nerftx_convert_kernel<<<(total + 255) / 256, 256, 0, stream>>>(
        W, ws + dstoff, K, N, Kpad);
  };

  // W_up padded 72->96 in K
  conv(W_up, 0, kEnc, kDim, kEncPad);
  for (int i = 0; i < 2; ++i) {
    const size_t lo = kWupSz + (size_t)i * kLayerSz;
    conv(Wq + (size_t)i * kDim * kDim, lo + kWqOff, kDim, kDim, kDim);
    conv(Wk + (size_t)i * kDim * kDim, lo + kWkOff, kDim, kDim, kDim);
    conv(Wv + (size_t)i * kDim * kDim, lo + kWvOff, kDim, kDim, kDim);
    conv(Wo + (size_t)i * kDim * kDim, lo + kWoOff, kDim, kDim, kDim);
    conv(W1 + (size_t)i * kDim * kFF,  lo + kW1Off, kDim, kFF, kDim);
    conv(W2 + (size_t)i * kFF * kDim,  lo + kW2Off, kFF, kDim, kFF);
  }

  nerftx_ray_kernel<<<n_rays, 128, 0, stream>>>(
      points, ws, b_up, ln1_s, ln1_b, ln2_s, ln2_b, b1, b2,
      w_pool, W_hit, b_hit, W_dc, b_dc, W_dv, b_dv,
      (float*)d_out, n_rays);
}
